// TriangleMultiplicativeModule_5411658793431
// MI455X (gfx1250) — matrix operand order repaired and checked by reading, not compiled, not run on hardware
//
#include <hip/hip_runtime.h>

typedef __bf16 bf16;
typedef __bf16 v16bf __attribute__((ext_vector_type(16)));
typedef float  v8f   __attribute__((ext_vector_type(8)));

#define LSEQ 512
#define DDIM 64
#define HDIM 64
#define NPOS (LSEQ * LSEQ)
#define EPSF 1e-5f

__device__ __forceinline__ v8f zero_v8f() {
  v8f z;
#pragma unroll
  for (int i = 0; i < 8; ++i) z[i] = 0.0f;
  return z;
}

__device__ __forceinline__ float sigmoidf_(float x) {
  return 1.0f / (1.0f + __expf(-x));
}

// A fragment for v_wmma_f32_16x16x32_bf16: 16(M) x 32(K) tile.
// src is row-major [16][lda] bf16 (K contiguous), kc = K-chunk base.
// Wave32 layout: lanes 0-15 -> M=lane, K groups {0..7,16..23}+0;
//                lanes 16-31 -> M=lane-16, K groups {8..15,24..31}.
__device__ __forceinline__ v16bf load_a(const bf16* src, int lda, int lane, int kc) {
  v16bf a;
  const int m = lane & 15;
  const int kb = kc + ((lane >> 4) << 3);
#pragma unroll
  for (int e = 0; e < 16; ++e) {
    int k = kb + (e & 7) + ((e >> 3) << 4);
    a[e] = src[m * lda + k];
  }
  return a;
}

// B fragment: 32(K) x 16(N) tile. src stored COLUMN-major: src[n*ldb + k]
// (K contiguous per column). Lanes 0-15: K={0..7,16..23}, lanes 16-31: K={8..15,24..31}, N=lane&15.
__device__ __forceinline__ v16bf load_b(const bf16* src, int ldb, int lane, int kc) {
  v16bf b;
  const int n = lane & 15;
  const int kb = kc + ((lane >> 4) << 3);
#pragma unroll
  for (int e = 0; e < 16; ++e) {
    b[e] = src[n * ldb + kb + (e & 7) + ((e >> 3) << 4)];
  }
  return b;
}

// ---------------------------------------------------------------------------
// Kernel A: per-position LayerNorm + 5 fused 64x64 projections (WMMA) +
// mask + sigmoid gates. Writes left/right channel-major bf16 for the einsum,
// and out_gate bf16 position-major for the epilogue kernel.
// Block: 128 threads (4 waves) handles 16 consecutive flattened positions.
// ---------------------------------------------------------------------------
__global__ __launch_bounds__(128) void tri_proj_kernel(
    const float* __restrict__ x, const int* __restrict__ seqmask,
    const float* __restrict__ norm_w, const float* __restrict__ norm_b,
    const float* __restrict__ left_w, const float* __restrict__ left_b,
    const float* __restrict__ right_w, const float* __restrict__ right_b,
    const float* __restrict__ lgate_w, const float* __restrict__ lgate_b,
    const float* __restrict__ rgate_w, const float* __restrict__ rgate_b,
    const float* __restrict__ ogate_w, const float* __restrict__ ogate_b,
    bf16* __restrict__ lt, bf16* __restrict__ rt, bf16* __restrict__ og) {
  __shared__ bf16 Wl[5][HDIM][DDIM];   // [mat][n][k], column-major (k contiguous)
  __shared__ bf16 xn[16][DDIM];        // normalized rows, bf16
  __shared__ float maskf[16];
  __shared__ bf16 bounce[2][HDIM][16]; // transpose bounce for left/right

  const int t = threadIdx.x;
  const int pos0 = blockIdx.x * 16;

  // Stage the 5 weight matrices column-major bf16 (L2-resident, 80 KB total).
  const float* Wsrc[5] = {left_w, right_w, lgate_w, rgate_w, ogate_w};
  for (int idx = t; idx < 5 * DDIM * HDIM; idx += 128) {
    int mi = idx >> 12;
    int r2 = idx & 4095;
    int k = r2 >> 6;
    int n = r2 & 63;
    Wl[mi][n][k] = (bf16)Wsrc[mi][k * HDIM + n];
  }

  // LayerNorm over D per position (threads 0..15, one row each).
  if (t < 16) {
    const int pos = pos0 + t;
    const float* xr = x + (size_t)pos * DDIM;
    float mu = 0.f, m2 = 0.f;
#pragma unroll
    for (int d = 0; d < DDIM; ++d) {
      float v = xr[d];
      mu += v;
      m2 += v * v;
    }
    mu *= (1.0f / DDIM);
    float var = m2 * (1.0f / DDIM) - mu * mu;
    float rs = rsqrtf(var + EPSF);
#pragma unroll
    for (int d = 0; d < DDIM; ++d)
      xn[t][d] = (bf16)(((xr[d] - mu) * rs) * norm_w[d] + norm_b[d]);
    const int i = pos >> 9, j = pos & (LSEQ - 1);
    maskf[t] = (float)(seqmask[i] * seqmask[j]);
  }
  __syncthreads();

  const int wv = t >> 5;     // wave id 0..3 -> N-tile
  const int lane = t & 31;

  const v16bf a0 = load_a(&xn[0][0], DDIM, lane, 0);
  const v16bf a1 = load_a(&xn[0][0], DDIM, lane, 32);

  v8f acc[5];
#pragma unroll
  for (int mi = 0; mi < 5; ++mi) {
    v8f c = zero_v8f();
    v16bf b0 = load_b(&Wl[mi][wv * 16][0], DDIM, lane, 0);
    c = __builtin_amdgcn_wmma_f32_16x16x32_bf16(false, a0, false, b0, (short)0, c,
                                                false, false);
    v16bf b1 = load_b(&Wl[mi][wv * 16][0], DDIM, lane, 32);
    c = __builtin_amdgcn_wmma_f32_16x16x32_bf16(false, a1, false, b1, (short)0, c,
                                                false, false);
    acc[mi] = c;
  }

  // Epilogue: bias + mask + sigmoid gates, out_gate store, transpose bounce.
  const int n = (wv << 4) + (lane & 15);
  const float lb = left_b[n], rb = right_b[n];
  const float lgb = lgate_b[n], rgb = rgate_b[n], ogb = ogate_b[n];
#pragma unroll
  for (int r = 0; r < 8; ++r) {
    int m = r + ((lane >> 4) << 3);
    float mk = maskf[m];
    float lv = (acc[0][r] + lb) * mk * sigmoidf_(acc[2][r] + lgb);
    float rv = (acc[1][r] + rb) * mk * sigmoidf_(acc[3][r] + rgb);
    float ov = sigmoidf_(acc[4][r] + ogb);
    og[(size_t)(pos0 + m) * HDIM + n] = (bf16)ov;
    bounce[0][n][m] = (bf16)lv;
    bounce[1][n][m] = (bf16)rv;
  }
  __syncthreads();

  // Channel-major writeout: lt/rt[ch][pos], 16 contiguous positions per row.
  {
    const int row = t & 63;
    const int which = t >> 6;  // 0 = left, 1 = right
    bf16* dst = (which == 0 ? lt : rt) + (size_t)row * NPOS + pos0;
#pragma unroll
    for (int q = 0; q < 16; ++q) dst[q] = bounce[which][row][q];
  }
}

// ---------------------------------------------------------------------------
// Kernel B: einsum out[i,j,d] = sum_k right[k,i,d]*left[k,j,d] as 64
// channel-wise GEMMs C_d(512x512) = R_d^T(512x512) * L_d(512x512), bf16 WMMA.
// Block: 128 threads (4 waves), 64x64 C tile; K slabs of 32 staged in LDS.
// ---------------------------------------------------------------------------
__global__ __launch_bounds__(128) void tri_einsum_kernel(
    const bf16* __restrict__ lt, const bf16* __restrict__ rt,
    float* __restrict__ oe) {
  __shared__ bf16 Asl[64][34];  // [m = i-i0][k], row-major (k contiguous)
  __shared__ bf16 Bsl[64][34];  // [n = j-j0][k], column-major (k contiguous)

  const int t = threadIdx.x;
  const int j0 = blockIdx.x * 64;
  const int i0 = blockIdx.y * 64;
  const int d = blockIdx.z;
  const bf16* Rd = rt + (size_t)d * NPOS;
  const bf16* Ld = lt + (size_t)d * NPOS;

  const int wv = t >> 5, lane = t & 31;
  v8f acc[4];
#pragma unroll
  for (int mt = 0; mt < 4; ++mt) acc[mt] = zero_v8f();

  const int kr = t >> 2;        // staging k-row: 0..31
  const int c0 = (t & 3) << 4;  // staging col base: 0,16,32,48

  for (int kk = 0; kk < LSEQ; kk += 32) {
    // A[m][k] = R[d][kk+k][i0+m] ; B[n][k] = L[d][kk+k][j0+n]
    const bf16* rrow = Rd + (size_t)(kk + kr) * LSEQ + i0 + c0;
    const bf16* lrow = Ld + (size_t)(kk + kr) * LSEQ + j0 + c0;
    if (kk + 32 < LSEQ) {  // prefetch next K slab (global_prefetch_b8)
      __builtin_prefetch(rrow + 32 * LSEQ, 0, 0);
      __builtin_prefetch(lrow + 32 * LSEQ, 0, 0);
    }
#pragma unroll
    for (int q = 0; q < 16; ++q) {
      Asl[c0 + q][kr] = rrow[q];
      Bsl[c0 + q][kr] = lrow[q];
    }
    __syncthreads();
    v16bf bfr = load_b(&Bsl[wv * 16][0], 34, lane, 0);
#pragma unroll
    for (int mt = 0; mt < 4; ++mt) {
      v16bf afr = load_a(&Asl[mt * 16][0], 34, lane, 0);
      acc[mt] = __builtin_amdgcn_wmma_f32_16x16x32_bf16(false, afr, false, bfr,
                                                        (short)0, acc[mt], false,
                                                        false);
    }
    __syncthreads();
  }

  // Store C channel-major f32: oe[d][i][j] (16 contiguous j per lane group).
  float* od = oe + (size_t)d * NPOS;
  const int n = j0 + (wv << 4) + (lane & 15);
#pragma unroll
  for (int mt = 0; mt < 4; ++mt) {
#pragma unroll
    for (int r = 0; r < 8; ++r) {
      int i = i0 + mt * 16 + r + ((lane >> 4) << 3);
      od[(size_t)i * LSEQ + n] = acc[mt][r];
    }
  }
}

// ---------------------------------------------------------------------------
// Kernel C: gather 64-ch vector per position from channel-major einsum out,
// LayerNorm over H, out-projection (WMMA) + bias, multiply by out_gate.
// Block: 128 threads (4 waves) per 16 positions.
// ---------------------------------------------------------------------------
__global__ __launch_bounds__(128) void tri_out_kernel(
    const float* __restrict__ oe, const float* __restrict__ onorm_w,
    const float* __restrict__ onorm_b, const float* __restrict__ out_w,
    const float* __restrict__ out_b, const bf16* __restrict__ og,
    float* __restrict__ out) {
  __shared__ bf16 Wo[DDIM][HDIM];   // [n][k] = out_w[k][n], column-major
  __shared__ float oelds[16][HDIM];
  __shared__ bf16 xn[16][HDIM];

  const int t = threadIdx.x;
  const int pos0 = blockIdx.x * 16;

  for (int idx = t; idx < HDIM * DDIM; idx += 128) {
    int k = idx >> 6, n = idx & 63;
    Wo[n][k] = (bf16)out_w[k * DDIM + n];
  }
  {
    const int m = t & 15;
    const int dbase = (t >> 4) << 3;
#pragma unroll
    for (int q = 0; q < 8; ++q) {
      int d = dbase + q;
      oelds[m][d] = oe[(size_t)d * NPOS + pos0 + m];
    }
  }
  __syncthreads();

  if (t < 16) {
    float mu = 0.f, m2 = 0.f;
#pragma unroll
    for (int h = 0; h < HDIM; ++h) {
      float v = oelds[t][h];
      mu += v;
      m2 += v * v;
    }
    mu *= (1.0f / HDIM);
    float var = m2 * (1.0f / HDIM) - mu * mu;
    float rs = rsqrtf(var + EPSF);
#pragma unroll
    for (int h = 0; h < HDIM; ++h)
      xn[t][h] = (bf16)(((oelds[t][h] - mu) * rs) * onorm_w[h] + onorm_b[h]);
  }
  __syncthreads();

  const int wv = t >> 5, lane = t & 31;
  const v16bf a0 = load_a(&xn[0][0], HDIM, lane, 0);
  const v16bf a1 = load_a(&xn[0][0], HDIM, lane, 32);
  const v16bf b0 = load_b(&Wo[wv * 16][0], HDIM, lane, 0);
  const v16bf b1 = load_b(&Wo[wv * 16][0], HDIM, lane, 32);
  v8f c = zero_v8f();
  c = __builtin_amdgcn_wmma_f32_16x16x32_bf16(false, a0, false, b0, (short)0, c,
                                              false, false);
  c = __builtin_amdgcn_wmma_f32_16x16x32_bf16(false, a1, false, b1, (short)0, c,
                                              false, false);

  const int n = (wv << 4) + (lane & 15);
  const float ob = out_b[n];
#pragma unroll
  for (int r = 0; r < 8; ++r) {
    int m = r + ((lane >> 4) << 3);
    size_t idx = (size_t)(pos0 + m) * DDIM + n;
    out[idx] = (c[r] + ob) * (float)og[idx];
  }
}

extern "C" void kernel_launch(void* const* d_in, const int* in_sizes, int n_in,
                              void* d_out, int out_size, void* d_ws,
                              size_t ws_size, hipStream_t stream) {
  (void)in_sizes; (void)n_in; (void)out_size; (void)ws_size;

  const float* x       = (const float*)d_in[0];
  const int*   sm      = (const int*)d_in[1];
  const float* norm_w  = (const float*)d_in[2];
  const float* norm_b  = (const float*)d_in[3];
  const float* left_w  = (const float*)d_in[4];
  const float* left_b  = (const float*)d_in[5];
  const float* right_w = (const float*)d_in[6];
  const float* right_b = (const float*)d_in[7];
  const float* lgate_w = (const float*)d_in[8];
  const float* lgate_b = (const float*)d_in[9];
  const float* rgate_w = (const float*)d_in[10];
  const float* rgate_b = (const float*)d_in[11];
  const float* ogate_w = (const float*)d_in[12];
  const float* ogate_b = (const float*)d_in[13];
  const float* onorm_w = (const float*)d_in[14];
  const float* onorm_b = (const float*)d_in[15];
  const float* out_w   = (const float*)d_in[16];
  const float* out_b   = (const float*)d_in[17];
  float* out = (float*)d_out;

  // Workspace layout (160 MB total):
  //   lt : bf16 [64][512][512]  channel-major left   ( 32 MB @ 0)
  //   rt : bf16 [64][512][512]  channel-major right  ( 32 MB @ 32MB)
  //   og : bf16 [512*512][64]   out_gate             ( 32 MB @ 64MB)
  //   oe : f32  [64][512][512]  einsum result        ( 64 MB @ 96MB)
  char* ws = (char*)d_ws;
  const size_t MB = (size_t)1 << 20;
  bf16*  lt = (bf16*)(ws);
  bf16*  rt = (bf16*)(ws + 32 * MB);
  bf16*  og = (bf16*)(ws + 64 * MB);
  float* oe = (float*)(ws + 96 * MB);

  tri_proj_kernel<<<NPOS / 16, 128, 0, stream>>>(
      x, sm, norm_w, norm_b, left_w, left_b, right_w, right_b, lgate_w,
      lgate_b, rgate_w, rgate_b, ogate_w, ogate_b, lt, rt, og);

  tri_einsum_kernel<<<dim3(LSEQ / 64, LSEQ / 64, HDIM), 128, 0, stream>>>(lt, rt,
                                                                          oe);

  tri_out_kernel<<<NPOS / 16, 128, 0, stream>>>(oe, onorm_w, onorm_b, out_w,
                                                out_b, og, out);
}
